// SequenceTrajectoryDiscriminator_50525995270862
// MI455X (gfx1250) — matrix operand order repaired and checked by reading, not compiled, not run on hardware
//
#include <hip/hip_runtime.h>
#include <hip/hip_bf16.h>

#define BB 512
#define TT 256
#define DD 64
#define HH 512
#define G3 1536

typedef __bf16 bf16;
typedef bf16 v16bf __attribute__((ext_vector_type(16)));
typedef float v8f  __attribute__((ext_vector_type(8)));
typedef unsigned int u32x4 __attribute__((ext_vector_type(4)));

// ---------------------------------------------------------------------------
// Fragment loaders per CDNA5 WMMA VGPR layouts (CDNA5 ISA §7.12.2)
// A (16-bit, 16x32 MxK): lanes 0-15 -> M=lane,    K={kb..kb+7} U {kb+16..kb+23}
//                        lanes 16-31 -> M=lane-16, K={kb+8..kb+15} U {kb+24..kb+31}
// B (16-bit, 32x16 KxN): lane -> N=lane%16, K same as A (table order)
// C/D (f32 16x16): vgpr v, lanes 0-15 -> (M=v, N=lane); lanes 16-31 -> (M=v+8)
// ---------------------------------------------------------------------------
__device__ __forceinline__ v16bf load_frag_a_bf16(const bf16* p) {
  v16bf a;
  ((u32x4*)&a)[0] = *(const u32x4*)(p);        // K block +0  (8 bf16 = 16B)
  ((u32x4*)&a)[1] = *(const u32x4*)(p + 16);   // K block +16
  return a;
}

__device__ __forceinline__ v16bf load_frag_a_f32(const float* p) {
  v16bf a;
#pragma unroll
  for (int i = 0; i < 8; ++i) a[i] = (bf16)p[i];
#pragma unroll
  for (int i = 0; i < 8; ++i) a[8 + i] = (bf16)p[16 + i];
  return a;
}

__device__ __forceinline__ v16bf load_frag_b_bf16(const bf16* p) {
  v16bf b;
  ((u32x4*)&b)[0] = *(const u32x4*)(p);        // K 0..7
  ((u32x4*)&b)[1] = *(const u32x4*)(p + 16);   // K block +16
  return b;
}

#define WMMA_BF16(A, Bf, C) \
  __builtin_amdgcn_wmma_f32_16x16x32_bf16(false, (A), false, (Bf), (short)0, (C), false, false)

// ---------------------------------------------------------------------------
__global__ void cvt_f32_bf16(const float* __restrict__ src, bf16* __restrict__ dst, int n) {
  int i = blockIdx.x * blockDim.x + threadIdx.x;
  if (i < n) dst[i] = (bf16)src[i];
}

// ---------------------------------------------------------------------------
// Fused persistent GRU + head. One workgroup owns 16 batch rows for the whole
// time loop (batch recurrences independent -> zero global sync). 512 threads =
// 16 wave32 waves; each wave owns 6 gate tiles with static gate mapping:
//   j=0,1 -> r   j=2,3 -> z   j=4,5 -> n
// K-outer / tile-inner: one LDS A fragment feeds 6 independent WMMA chains.
// An opaque 32-bit zero ('tz', redefined per timestep via empty asm) is added
// to the per-tile element offsets: this defeats LICM (round-2 spill blowup)
// while keeping loads derived from the kernel-arg pointers, so they lower to
// global_load_b128 (saddr + voffset) instead of flat_load (round-3 regression).
// ---------------------------------------------------------------------------
__global__ __launch_bounds__(512, 1)
void gru_fused_kernel(const float* __restrict__ features,  // [B,T,D] f32
                      const bf16*  __restrict__ whh,       // [3H,H] bf16
                      const bf16*  __restrict__ wih,       // [3H,D] bf16
                      const float* __restrict__ b_ih,      // [3H]
                      const float* __restrict__ b_hh,      // [3H]
                      const bf16*  __restrict__ w1b,       // [H,H] bf16
                      const float* __restrict__ b1,        // [H]
                      const float* __restrict__ w2,        // [H]
                      const float* __restrict__ b2,        // [1]
                      float* __restrict__ out)             // [B]
{
  __shared__ float rsum[16][HH];        // 32 KB sigmoid arg for r
  __shared__ float zsum[16][HH];        // 32 KB sigmoid arg for z
  __shared__ float xnl [16][HH];        // 32 KB x-side of n gate
  __shared__ float hnl [16][HH];        // 32 KB h-side of n gate
  __shared__ bf16  hbf [16][HH + 8];    // 16.25 KB, +8 pad -> 4-bank row skew

  const int tid   = threadIdx.x;
  const int lane  = tid & 31;
  const int wave  = tid >> 5;           // 0..15
  const int bBase = blockIdx.x * 16;

  for (int i = tid; i < 16 * (HH + 8); i += 512) (&hbf[0][0])[i] = (bf16)0.0f;
  __syncthreads();

  const int nlo   = lane & 15;
  const int mOff  = (lane < 16) ? 0 : 8;
  const int aSel  = (lane < 16) ? 0 : 8;    // A-frag K sub-block per lane half
  const int bSel  = (lane < 16) ? 0 : 8;    // B-frag K sub-block per lane half

  // ---- loop-invariant per-tile state (constant across all 256 steps) -------
  int   offH[6];        // element offset of w_hh fragment row per tile
  int   offX[6];        // element offset of w_ih fragment row per tile
  int   colL[6];        // column within gate segment [0,512)
  float biasS[4];       // b_ih+b_hh for r/z tiles
  float biasXn[2], biasHn[2];
#pragma unroll
  for (int j = 0; j < 6; ++j) {
    const int col = (wave + 16 * j) * 16 + nlo;   // this lane's gate column
    offH[j] = col * HH + bSel;
    offX[j] = col * DD + bSel;
    colL[j] = col & (HH - 1);
    if (j < 4)      biasS[j]      = b_ih[col] + b_hh[col];
    else          { biasXn[j - 4] = b_ih[col];  biasHn[j - 4] = b_hh[col]; }
  }
  const bf16* aLds = &hbf[0][0] + (size_t)nlo * (HH + 8) + aSel;

  for (int t = 0; t < TT; ++t) {
    // Opaque zero, redefined every timestep: offsets formally change per t,
    // so weight loads stay inside the loop (no LICM -> no scratch spill),
    // while staying derived from kernel args (address space 1 -> global_load).
    int tz = 0;
    asm volatile("" : "+s"(tz));

    // Per-timestep tile fragment bases (6 cheap address adds per step).
    const bf16* wbH_t[6];
    const bf16* wbX_t[6];
#pragma unroll
    for (int j = 0; j < 6; ++j) {
      wbH_t[j] = whh + (offH[j] + tz);
      wbX_t[j] = wih + (offX[j] + tz);
    }

    const float* xrow = features + (size_t)bBase * (TT * DD) + (size_t)t * DD
                        + (size_t)nlo * (TT * DD) + aSel;

    // ---- recurrent GEMM: K-outer, 6 independent accumulator chains ---------
    v8f acc[6] = {};
#pragma unroll 4
    for (int kb = 0; kb < HH; kb += 32) {
      v16bf a = load_frag_a_bf16(aLds + kb);          // one LDS A frag, 6 uses
#pragma unroll
      for (int j = 0; j < 6; ++j) {
        v16bf bb = load_frag_b_bf16(wbH_t[j] + kb);   // kb folds into imm24
        acc[j] = WMMA_BF16(a, bb, acc[j]);
      }
    }

    // ---- input projection (K=64): r/z fold into same accumulators ----------
    v8f accx[2] = {};
#pragma unroll
    for (int kb = 0; kb < DD; kb += 32) {
      v16bf a = load_frag_a_f32(xrow + kb);
#pragma unroll
      for (int j = 0; j < 4; ++j) {
        v16bf bb = load_frag_b_bf16(wbX_t[j] + kb);
        acc[j] = WMMA_BF16(a, bb, acc[j]);            // x+h fused for r,z
      }
#pragma unroll
      for (int j = 4; j < 6; ++j) {
        v16bf bb = load_frag_b_bf16(wbX_t[j] + kb);
        accx[j - 4] = WMMA_BF16(a, bb, accx[j - 4]);  // n gate: x kept apart
      }
    }

    // ---- stage gate pre-activations in LDS ---------------------------------
#pragma unroll
    for (int v = 0; v < 8; ++v) {
      rsum[mOff + v][colL[0]] = acc[0][v] + biasS[0];
      rsum[mOff + v][colL[1]] = acc[1][v] + biasS[1];
      zsum[mOff + v][colL[2]] = acc[2][v] + biasS[2];
      zsum[mOff + v][colL[3]] = acc[3][v] + biasS[3];
      hnl [mOff + v][colL[4]] = acc[4][v] + biasHn[0];
      hnl [mOff + v][colL[5]] = acc[5][v] + biasHn[1];
      xnl [mOff + v][colL[4]] = accx[0][v] + biasXn[0];
      xnl [mOff + v][colL[5]] = accx[1][v] + biasXn[1];
    }
    __syncthreads();

    // ---- gate phase: thread 'tid' owns column tid for all 16 rows ----------
    {
      const int jc = tid;
#pragma unroll 4
      for (int i = 0; i < 16; ++i) {
        float r  = 1.0f / (1.0f + __expf(-rsum[i][jc]));
        float z  = 1.0f / (1.0f + __expf(-zsum[i][jc]));
        float nv = tanhf(xnl[i][jc] + r * hnl[i][jc]);
        float ho = (float)hbf[i][jc];
        hbf[i][jc] = (bf16)((1.0f - z) * nv + z * ho);
      }
    }
    __syncthreads();
  }

  // ---- head: y = relu(h @ w1^T + b1) via WMMA (2 tiles per wave) -----------
  {
    const bf16* wb0 = w1b + (size_t)((wave)      * 16 + nlo) * HH + bSel;
    const bf16* wb1 = w1b + (size_t)((wave + 16) * 16 + nlo) * HH + bSel;
    v8f acc0 = {}, acc1 = {};
#pragma unroll 4
    for (int kb = 0; kb < HH; kb += 32) {
      v16bf a = load_frag_a_bf16(aLds + kb);
      acc0 = WMMA_BF16(a, load_frag_b_bf16(wb0 + kb), acc0);
      acc1 = WMMA_BF16(a, load_frag_b_bf16(wb1 + kb), acc1);
    }
    const int c0 = wave * 16 + nlo, c1 = c0 + 256;
    const float bb0 = b1[c0], bb1 = b1[c1];
#pragma unroll
    for (int v = 0; v < 8; ++v) {
      float y0 = acc0[v] + bb0;
      float y1 = acc1[v] + bb1;
      rsum[mOff + v][c0] = (y0 > 0.0f) ? y0 : 0.0f;   // stage y in LDS
      rsum[mOff + v][c1] = (y1 > 0.0f) ? y1 : 0.0f;
    }
  }
  __syncthreads();

  // ---- out[b] = dot(y[b,:], w2) + b2 : wave 'w' owns batch row 'w' ---------
  {
    float s = 0.0f;
    for (int jj = lane; jj < HH; jj += 32) s += rsum[wave][jj] * w2[jj];
#pragma unroll
    for (int off = 16; off >= 1; off >>= 1) s += __shfl_xor(s, off, 32);
    if (lane == 0) out[bBase + wave] = s + b2[0];
  }
}

// ---------------------------------------------------------------------------
extern "C" void kernel_launch(void* const* d_in, const int* in_sizes, int n_in,
                              void* d_out, int out_size, void* d_ws, size_t ws_size,
                              hipStream_t stream) {
  const float* features = (const float*)d_in[0];  // [B,T,D]
  const float* w_ih     = (const float*)d_in[1];  // [3H,D]
  const float* w_hh     = (const float*)d_in[2];  // [3H,H]
  const float* b_ih     = (const float*)d_in[3];  // [3H]
  const float* b_hh     = (const float*)d_in[4];  // [3H]
  const float* w1       = (const float*)d_in[5];  // [H,H]
  const float* b1       = (const float*)d_in[6];  // [H]
  const float* w2       = (const float*)d_in[7];  // [1,H]
  const float* b2       = (const float*)d_in[8];  // [1]
  float* out = (float*)d_out;

  // Workspace layout (bf16 weight copies, ~2.3 MB)
  char* ws = (char*)d_ws;
  bf16* whh_bf = (bf16*)(ws);
  bf16* wih_bf = (bf16*)(ws + (size_t)(3 * HH * HH) * sizeof(bf16));
  bf16* w1_bf  = (bf16*)(ws + (size_t)(3 * HH * HH + 3 * HH * DD) * sizeof(bf16));

  const int nWhh = 3 * HH * HH;   // 786432
  const int nWih = 3 * HH * DD;   // 98304
  const int nW1  = HH * HH;       // 262144

  cvt_f32_bf16<<<(nWhh + 255) / 256, 256, 0, stream>>>(w_hh, whh_bf, nWhh);
  cvt_f32_bf16<<<(nWih + 255) / 256, 256, 0, stream>>>(w_ih, wih_bf, nWih);
  cvt_f32_bf16<<<(nW1  + 255) / 256, 256, 0, stream>>>(w1,   w1_bf,  nW1);

  gru_fused_kernel<<<BB / 16, 512, 0, stream>>>(
      features, whh_bf, wih_bf, b_ih, b_hh, w1_bf, b1, w2, b2, out);
}
